// EmbeddingRNNClassifier_62483184222211
// MI455X (gfx1250) — matrix operand order repaired and checked by reading, not compiled, not run on hardware
//
#include <hip/hip_runtime.h>
#include <hip/hip_bf16.h>
#include <math.h>

// Problem dims
#define BATCH 64
#define SEQ   512
#define EMB   512
#define HID   1024
#define NOUT  3

typedef __bf16 v16bf __attribute__((ext_vector_type(16)));
typedef __bf16 v8bf  __attribute__((ext_vector_type(8)));
typedef float  v8f   __attribute__((ext_vector_type(8)));
typedef unsigned int u32x4 __attribute__((ext_vector_type(4)));
typedef int   i32x8 __attribute__((ext_vector_type(8)));
typedef int   i32x4 __attribute__((ext_vector_type(4)));

__device__ __forceinline__ v8f zero_v8f() {
    v8f z = {0.f,0.f,0.f,0.f,0.f,0.f,0.f,0.f};
    return z;
}

__device__ __forceinline__ v8f wmma_bf16(v16bf a, v16bf b, v8f c) {
    // D = A(16x32 bf16) * B(32x16 bf16) + C(16x16 f32)
    return __builtin_amdgcn_wmma_f32_16x16x32_bf16(
        false, a, false, b, (short)0, c, false, false);
}

// A fragment (16x32 bf16, row-major source).
// lanes 0-15: m=lane, half=0; lanes 16-31: m=lane-16, half=1
// a[0..7]  = row[k + 8*half + 0..7]; a[8..15] = row[k + 16 + 8*half + 0..7]
__device__ __forceinline__ v16bf load_fragA(const __bf16* rowp, int kbase) {
    v8bf lo = *(const v8bf*)(rowp + kbase);
    v8bf hi = *(const v8bf*)(rowp + kbase + 16);
    return __builtin_shufflevector(lo, hi, 0,1,2,3,4,5,6,7,8,9,10,11,12,13,14,15);
}

// B fragment (32x16): per lane n = lane&15, b[0..15] = Wrow_n[k + 16*half + 0..15]
__device__ __forceinline__ v16bf load_fragB(const __bf16* wrow, int kbase) {
    v8bf lo = *(const v8bf*)(wrow + kbase);
    v8bf hi = *(const v8bf*)(wrow + kbase + 8);
    return __builtin_shufflevector(lo, hi, 0,1,2,3,4,5,6,7,8,9,10,11,12,13,14,15);
}

// ---------------------------------------------------------------- prep kernels

__global__ __launch_bounds__(256)
void rnn_gather_embed(const int* __restrict__ seq, const float* __restrict__ emb,
                      __bf16* __restrict__ X) {
    const int bt  = blockIdx.x;
    const int row = seq[bt];
    const float* src = emb + (size_t)row * EMB;
    __bf16* dst = X + (size_t)bt * EMB;
    for (int e = threadIdx.x; e < EMB; e += blockDim.x)
        dst[e] = (__bf16)src[e];
}

__global__ __launch_bounds__(256)
void rnn_cvt_bf16(const float* __restrict__ src, __bf16* __restrict__ dst, int n) {
    int i = blockIdx.x * blockDim.x + threadIdx.x;
    if (i < n) dst[i] = (__bf16)src[i];
}

__global__ __launch_bounds__(256)
void rnn_init_state(__bf16* __restrict__ h0, __bf16* __restrict__ h1, int n,
                    int* __restrict__ cnt) {
    int i = blockIdx.x * blockDim.x + threadIdx.x;
    if (i < n) { h0[i] = (__bf16)0.0f; h1[i] = (__bf16)0.0f; }
    if (i == 0) *cnt = 0;
}

// ------------------------------------------------- phase 1: input projection
// XP stored [T][B][H]: XP[(t*64 + b)*1024 + h]
__global__ __launch_bounds__(256)
void rnn_xproj(const __bf16* __restrict__ X, const __bf16* __restrict__ W,
               const float* __restrict__ bias, float* __restrict__ XP) {
    const int lane = threadIdx.x & 31;
    const int wave = threadIdx.x >> 5;
    const int gw   = blockIdx.x * 8 + wave;
    const int mt   = gw & 2047;
    const int ng   = gw >> 11;
    const int half = lane >> 4;
    const int lm   = lane & 15;

    const __bf16* arow  = X + (size_t)(mt * 16 + lm) * EMB;
    const __bf16* wrow0 = W + (size_t)(ng * 64 +  0 + lm) * EMB;
    const __bf16* wrow1 = W + (size_t)(ng * 64 + 16 + lm) * EMB;
    const __bf16* wrow2 = W + (size_t)(ng * 64 + 32 + lm) * EMB;
    const __bf16* wrow3 = W + (size_t)(ng * 64 + 48 + lm) * EMB;

    v8f acc0 = zero_v8f(), acc1 = zero_v8f(), acc2 = zero_v8f(), acc3 = zero_v8f();

    for (int k = 0; k < EMB; k += 32) {
        v16bf a  = load_fragB(arow, k + 16 * half);
        v16bf b0 = load_fragB(wrow0, k + 16 * half);
        v16bf b1 = load_fragB(wrow1, k + 16 * half);
        v16bf b2 = load_fragB(wrow2, k + 16 * half);
        v16bf b3 = load_fragB(wrow3, k + 16 * half);
        acc0 = wmma_bf16(a, b0, acc0);
        acc1 = wmma_bf16(a, b1, acc1);
        acc2 = wmma_bf16(a, b2, acc2);
        acc3 = wmma_bf16(a, b3, acc3);
    }

    const int c0 = ng * 64 + lm;
    const float bv0 = bias[c0], bv1 = bias[c0 + 16], bv2 = bias[c0 + 32], bv3 = bias[c0 + 48];
#pragma unroll
    for (int r = 0; r < 8; ++r) {
        const int row = mt * 16 + r + 8 * half;   // row = b*SEQ + t
        const int b  = row >> 9;
        const int tt = row & (SEQ - 1);
        float* o = XP + ((size_t)tt * BATCH + b) * HID + c0;
        o[ 0] = acc0[r] + bv0;
        o[16] = acc1[r] + bv1;
        o[32] = acc2[r] + bv2;
        o[48] = acc3[r] + bv3;
    }
}

// -------------------------------------------------- phase 2: recurrence (persistent)
// 16 blocks x 8 waves. Block owns 64 consecutive output columns; its W_hh slice
// (64 x 1024 bf16 = 128KB) is staged once in LDS (padded rows) and reused for all
// 512 timesteps. Wave = (mt = wave&3, sub = wave>>2) -> 2 N-tiles, shared A frag.
#define RNN_BLOCKS 16
#define WSTRIDE    1032   // 1024 + 2*4 bf16: 2-dword pad after each 512-elem chunk

__device__ __forceinline__ int lds_kidx(int koff) {
    return koff + (koff >= 512 ? 4 : 0);
}

__global__ __launch_bounds__(256)
void rnn_recur(const __bf16* __restrict__ Whh, const float* __restrict__ bhh,
               const float* __restrict__ XP, __bf16* __restrict__ hA,
               __bf16* __restrict__ hB, float* __restrict__ hfinal,
               int* __restrict__ cnt) {
    __shared__ __bf16 Wl[64 * WSTRIDE];          // 132,096 bytes

    const int lane = threadIdx.x & 31;
    const int wave = threadIdx.x >> 5;
    const int half = lane >> 4;
    const int lm   = lane & 15;
    const int mt   = wave & 3;                   // batch tile
    const int sub  = wave >> 2;                  // 0..1: which 32-col group
    const int base_col = blockIdx.x * 64;

#if defined(__gfx1250__) && __has_builtin(__builtin_amdgcn_tensor_load_to_lds) && __has_builtin(__builtin_amdgcn_s_wait_tensorcnt)
    // One-shot TDM prefetch of this block's W slice into LDS (2D tile 64 x 1024,
    // data_size=2B, LDS padding: 2 dwords after every 256 dwords -> WSTRIDE rows).
    if (wave == 0) {
        unsigned long long ga = (unsigned long long)(const void*)(Whh + (size_t)base_col * HID);
        unsigned lds_off = (unsigned)(size_t)(void*)&Wl[0];
        u32x4 g0;
        g0[0] = 1u;                                   // count=1, user descriptor
        g0[1] = lds_off;                              // lds_addr
        g0[2] = (unsigned)(ga & 0xFFFFFFFFull);       // global_addr[31:0]
        g0[3] = (unsigned)((ga >> 32) & 0x1FFFFFFull) | (2u << 30);  // addr[56:32] | type=2
        i32x8 g1;
        g1[0] = (1 << 16)            // data_size = 2 bytes
              | (1 << 20)            // pad_enable
              | (7 << 22)            // pad_interval: every 256 dwords
              | (1 << 25);           // pad_amount: 2 dwords
        g1[1] = (int)(1024u << 16);  // tensor_dim0 = 1024 (lo16 into [63:48])
        g1[2] = (int)(64u << 16);    // tensor_dim1 = 64
        g1[3] = (int)(1024u << 16);  // tile_dim0 = 1024
        g1[4] = 64;                  // tile_dim1 = 64 (tile_dim2 = 0)
        g1[5] = 1024;                // tensor_dim0_stride = 1024
        g1[6] = 0;
        g1[7] = 0;
        i32x4 g2 = {0, 0, 0, 0};
        i32x4 g3 = {0, 0, 0, 0};
        i32x8 g4 = {0, 0, 0, 0, 0, 0, 0, 0};
        __builtin_amdgcn_tensor_load_to_lds(g0, g1, g2, g3, g4, 0);
        __builtin_amdgcn_s_wait_tensorcnt(0);
    }
    __syncthreads();
#endif
    // Cooperative staging (authoritative data; same padded layout).
    for (int idx = threadIdx.x; idx < 64 * (HID / 8); idx += 256) {
        const int c  = idx >> 7;        // local col 0..63
        const int ck = idx & 127;       // 8-elem chunk
        const int k  = ck * 8;
        *(v8bf*)&Wl[c * WSTRIDE + lds_kidx(k)] =
            *(const v8bf*)(Whh + (size_t)(base_col + c) * HID + k);
    }
    __syncthreads();

    const __bf16* w0 = Wl + (size_t)(sub * 32 + lm) * WSTRIDE;
    const __bf16* w1 = Wl + (size_t)(sub * 32 + 16 + lm) * WSTRIDE;
    const int col0 = base_col + sub * 32 + lm;
    const int col1 = col0 + 16;
    const float bv0 = bhh[col0], bv1 = bhh[col1];

    for (int t = 0; t < SEQ; ++t) {
        const __bf16* hin  = (t & 1) ? hB : hA;
        __bf16*       hout = (t & 1) ? hA : hB;
        const __bf16* arow = hin + (size_t)(mt * 16 + lm) * HID;

        v8f acc0 = zero_v8f(), acc1 = zero_v8f();
        for (int k = 0; k < HID; k += 32) {
            const int koff = k + 16 * half;
            v16bf a  = load_fragB(arow, k + 16 * half);
            v16bf b0 = load_fragB(w0, lds_kidx(koff));
            v16bf b1 = load_fragB(w1, lds_kidx(koff));
            acc0 = wmma_bf16(a, b0, acc0);
            acc1 = wmma_bf16(a, b1, acc1);
        }

        const float* xpt = XP + (size_t)t * BATCH * HID;
#pragma unroll
        for (int r = 0; r < 8; ++r) {
            const int b = mt * 16 + r + 8 * half;
            const float* xp = xpt + (size_t)b * HID;
            const float v0 = tanhf(acc0[r] + xp[col0] + bv0);
            const float v1 = tanhf(acc1[r] + xp[col1] + bv1);
            hout[(size_t)b * HID + col0] = (__bf16)v0;
            hout[(size_t)b * HID + col1] = (__bf16)v1;
            if (t == SEQ - 1) {
                hfinal[(size_t)b * HID + col0] = v0;
                hfinal[(size_t)b * HID + col1] = v1;
            }
        }

        if (t + 1 < SEQ)  // warm L2/WGP$ with next step's xp slab
            __builtin_prefetch(XP + ((size_t)(t + 1) * BATCH + mt * 16 + lm) * HID + col0, 0, 1);

        // ---- grid-wide barrier (monotonic counter, release/acquire) ----
        __threadfence();                 // flush this block's h stores (device scope)
        __syncthreads();
        if (threadIdx.x == 0) {
            __hip_atomic_fetch_add(cnt, 1, __ATOMIC_RELEASE, __HIP_MEMORY_SCOPE_AGENT);
            const int target = (t + 1) * RNN_BLOCKS;
            while (__hip_atomic_load(cnt, __ATOMIC_ACQUIRE, __HIP_MEMORY_SCOPE_AGENT) < target) {
                __builtin_amdgcn_s_sleep(1);
            }
        }
        __syncthreads();
    }
}

// -------------------------------------------------- phase 3: head + log_softmax(axis=0)
__global__ __launch_bounds__(192)
void rnn_head(const float* __restrict__ hf, const float* __restrict__ Who,
              const float* __restrict__ bho, float* __restrict__ out) {
    __shared__ float lg[BATCH * NOUT];
    __shared__ float lse[NOUT];
    const int tid = threadIdx.x;
    const int b = tid / NOUT, o = tid % NOUT;

    float s = bho[o];
    const float* hr = hf + (size_t)b * HID;
    const float* wr = Who + (size_t)o * HID;
    for (int i = 0; i < HID; ++i) s += hr[i] * wr[i];
    lg[tid] = s;
    __syncthreads();

    if (tid < NOUT) {
        float m = -3.4e38f;
        for (int bb = 0; bb < BATCH; ++bb) m = fmaxf(m, lg[bb * NOUT + tid]);
        float se = 0.f;
        for (int bb = 0; bb < BATCH; ++bb) se += expf(lg[bb * NOUT + tid] - m);
        lse[tid] = m + logf(se);
    }
    __syncthreads();
    out[tid] = lg[tid] - lse[o];
}

// ---------------------------------------------------------------- launch

extern "C" void kernel_launch(void* const* d_in, const int* in_sizes, int n_in,
                              void* d_out, int out_size, void* d_ws, size_t ws_size,
                              hipStream_t stream) {
    const int*   seq = (const int*)d_in[0];
    const float* emb = (const float*)d_in[1];
    const float* Wih = (const float*)d_in[2];
    const float* bih = (const float*)d_in[3];
    const float* Whh = (const float*)d_in[4];
    const float* bhh = (const float*)d_in[5];
    const float* Who = (const float*)d_in[6];
    const float* bho = (const float*)d_in[7];
    float* out = (float*)d_out;

    char* p = (char*)d_ws;
    auto take = [&](size_t bytes) -> char* {
        char* r = p;
        p += (bytes + 255) & ~(size_t)255;
        return r;
    };
    __bf16* X    = (__bf16*)take((size_t)BATCH * SEQ * EMB * 2);  // 32 MB
    __bf16* Wihb = (__bf16*)take((size_t)HID * EMB * 2);          // 1 MB
    __bf16* Whhb = (__bf16*)take((size_t)HID * HID * 2);          // 2 MB
    float*  XP   = (float*)take((size_t)BATCH * SEQ * HID * 4);   // 128 MB, [T][B][H]
    __bf16* h0   = (__bf16*)take((size_t)BATCH * HID * 2);
    __bf16* h1   = (__bf16*)take((size_t)BATCH * HID * 2);
    float*  hf   = (float*)take((size_t)BATCH * HID * 4);
    int*    cnt  = (int*)take(256);

    rnn_gather_embed<<<BATCH * SEQ, 256, 0, stream>>>(seq, emb, X);
    rnn_cvt_bf16<<<(HID * EMB + 255) / 256, 256, 0, stream>>>(Wih, Wihb, HID * EMB);
    rnn_cvt_bf16<<<(HID * HID + 255) / 256, 256, 0, stream>>>(Whh, Whhb, HID * HID);
    rnn_init_state<<<(BATCH * HID + 255) / 256, 256, 0, stream>>>(h0, h1, BATCH * HID, cnt);

    rnn_xproj<<<4096, 256, 0, stream>>>(X, Wihb, bih, XP);
    rnn_recur<<<RNN_BLOCKS, 256, 0, stream>>>(Whhb, bhh, XP, h0, h1, hf, cnt);
    rnn_head<<<1, 192, 0, stream>>>(hf, Who, bho, out);
}
